// StableVectorQuantizer_73890617361026
// MI455X (gfx1250) — compile-verified, matrix operand order checked by reading, not run on hardware
//
#include <hip/hip_runtime.h>
#include <hip/hip_bf16.h>

typedef __attribute__((ext_vector_type(16))) __bf16 v16bf;
typedef __attribute__((ext_vector_type(8)))  __bf16 v8bf;
typedef __attribute__((ext_vector_type(8)))  float  v8f;

#define N_TOK   65536
#define DIM     64
#define K_CODES 1024
#define QUANT_N (N_TOK * DIM)

// workspace byte offsets
#define WS_CBBF   0                 // __bf16[K_CODES*DIM]  = 131072 B
#define WS_ESQ    131072            // float[K_CODES]       =   4096 B
#define WS_CNT    135168            // float[K_CODES]       =   4096 B
#define WS_LOSS   139264            // float[1]

// ---------------------------------------------------------------------------
// Prep: codebook f32 -> bf16, ||e||^2, zero counts & loss accumulator.
// ---------------------------------------------------------------------------
__global__ void vq_prep(const float* __restrict__ cb,
                        __bf16* __restrict__ cb_bf,
                        float* __restrict__ esq,
                        float* __restrict__ counts,
                        float* __restrict__ loss) {
    int c = blockIdx.x * blockDim.x + threadIdx.x;
    if (c < K_CODES) {
        const float* row = cb + c * DIM;
        float s = 0.0f;
#pragma unroll
        for (int d = 0; d < DIM; ++d) {
            float v = row[d];
            s += v * v;
            cb_bf[c * DIM + d] = (__bf16)v;
        }
        esq[c]    = s;
        counts[c] = 0.0f;
        if (c == 0) *loss = 0.0f;
    }
}

// ---------------------------------------------------------------------------
// Main: per-wave 16-token tile vs full 1024-code book via bf16 WMMA.
// Codebook (bf16) + norms staged in LDS (132 KB of the 320 KB WGP LDS).
// ---------------------------------------------------------------------------
__global__ void __launch_bounds__(256)
vq_main(const float* __restrict__ x,
        const __bf16* __restrict__ cb_bf,
        const float* __restrict__ esq_g,
        const float* __restrict__ cb_f32,
        float* __restrict__ out_q,
        float* __restrict__ out_idx,
        float* __restrict__ counts,
        float* __restrict__ loss) {
    extern __shared__ char smem[];
    __bf16* cb_lds  = (__bf16*)smem;                          // 131072 B
    float*  esq_lds = (float*)(smem + 131072);                //   4096 B
    int*    idx_lds = (int*)(smem + 131072 + 4096);           //    512 B (8 waves x 16)
    float*  red     = (float*)(smem + 131072 + 4096 + 512);   //     32 B

    const int tid  = threadIdx.x;
    const int lane = tid & 31;
    const int warp = tid >> 5;
    const int half = lane >> 4;      // 0 or 1
    const int n    = lane & 15;      // column / row-in-tile id

    // ---- stage bf16 codebook into LDS (128 KB) ------------------------------
#if defined(__gfx1250__)
    {
        // CDNA5 async copy global->LDS (VGLOBAL op, ASYNCcnt-tracked), GV form:
        //   global_load_async_to_lds_b128  vdst(=LDS byte addr), v[addr], off
        unsigned lds0 = (unsigned)(unsigned long long)
            (__attribute__((address_space(3))) char*)smem;   // 32-bit LDS offset
        unsigned long long g0 = (unsigned long long)cb_bf;
        const int nchunk = (K_CODES * DIM * 2) / 16;         // 8192 x 16B
        for (int i = tid; i < nchunk; i += 256) {
            unsigned           laddr = lds0 + (unsigned)i * 16u;
            unsigned long long gaddr = g0 + (unsigned long long)i * 16ull;
            asm volatile("global_load_async_to_lds_b128 %0, %1, off"
                         :: "v"(laddr), "v"(gaddr) : "memory");
        }
        asm volatile("s_wait_asynccnt 0x0" ::: "memory");
    }
#else
    {
        const uint4* src = (const uint4*)cb_bf;
        uint4*       dst = (uint4*)cb_lds;
        const int nchunk = (K_CODES * DIM * 2) / 16;   // 8192
#pragma unroll 4
        for (int i = tid; i < nchunk; i += 256) dst[i] = src[i];
    }
#endif
    for (int i = tid; i < K_CODES; i += 256) esq_lds[i] = esq_g[i];
    __syncthreads();

    const int tile  = blockIdx.x * 8 + warp;   // 4096 wave tiles
    const int tbase = tile * 16;               // first token of this tile

    // ---- A fragments (16 tokens x 64 dims, two 16x32 bf16 frags) + ||x||^2
    // ISA 7.12.2 16-bit A layout: lanes 0-15 hold K {b*8+0..7, 16+b*8+0..7}, b=half
    v16bf a0, a1;
    float part = 0.0f;
    {
        const float* row = x + (size_t)(tbase + n) * DIM;
        const int kb = half * 8;
#pragma unroll
        for (int i = 0; i < 8; ++i) {
            float f0 = row[kb + i];
            float f1 = row[16 + kb + i];
            float f2 = row[32 + kb + i];
            float f3 = row[48 + kb + i];
            a0[i]     = (__bf16)f0;  a0[8 + i] = (__bf16)f1;
            a1[i]     = (__bf16)f2;  a1[8 + i] = (__bf16)f3;
            part += f0 * f0 + f1 * f1 + f2 * f2 + f3 * f3;
        }
    }
    // lanes L and L+16 jointly cover all 64 dims of token (L&15)
    const float xsq = part + __shfl_xor(part, 16, 32);

    // Hoist the loop-invariant ||x_M||^2 broadcasts OUT of the code-tile loop:
    // C layout puts (M = v + 8*half, N = n) in this lane -> need xsq of lane v+8*half.
    float xs_row[8];
#pragma unroll
    for (int v = 0; v < 8; ++v) xs_row[v] = __shfl(xsq, v + (half << 3), 32);

    float best_d[8];
    int   best_i[8];
#pragma unroll
    for (int v = 0; v < 8; ++v) { best_d[v] = 3.4e38f; best_i[v] = 0; }

    // ---- sweep 64 code tiles of 16 codes each (unroll 2: two independent
    //      WMMA accumulation chains in flight) -------------------------------
#pragma unroll 2
    for (int j = 0; j < K_CODES / 16; ++j) {
        // B = E^T: B-fragment columns load exactly like A rows of E (row-major)
        const __bf16* brow = cb_lds + (j * 16 + n) * DIM + half * 8;  // 16B aligned
        v16bf b0, b1;
        {
            v8bf lo0 = *(const v8bf*)(brow);
            v8bf hi0 = *(const v8bf*)(brow + 16);
            v8bf lo1 = *(const v8bf*)(brow + 32);
            v8bf hi1 = *(const v8bf*)(brow + 48);
#pragma unroll
            for (int i = 0; i < 8; ++i) {
                b0[i] = lo0[i]; b0[8 + i] = hi0[i];
                b1[i] = lo1[i]; b1[8 + i] = hi1[i];
            }
        }

        v8f c = {};
        c = __builtin_amdgcn_wmma_f32_16x16x32_bf16(false, a0, false, b0,
                                                    (short)0, c, false, false);
        c = __builtin_amdgcn_wmma_f32_16x16x32_bf16(false, a1, false, b1,
                                                    (short)0, c, false, false);

        const float en  = esq_lds[j * 16 + n];
        const int   idx = j * 16 + n;
#pragma unroll
        for (int v = 0; v < 8; ++v) {
            float d = (xs_row[v] + en) - 2.0f * c[v];
            if (d < best_d[v]) { best_d[v] = d; best_i[v] = idx; }
        }
    }

    // ---- argmin butterfly across the 16 lanes sharing each token row
    //      (xor 1,2,4,8 stays inside each 16-lane half of the C matrix)
#pragma unroll
    for (int off = 1; off < 16; off <<= 1) {
#pragma unroll
        for (int v = 0; v < 8; ++v) {
            float od = __shfl_xor(best_d[v], off, 32);
            int   oi = __shfl_xor(best_i[v], off, 32);
            if (od < best_d[v] || (od == best_d[v] && oi < best_i[v])) {
                best_d[v] = od; best_i[v] = oi;
            }
        }
    }

    if (n == 0) {  // lanes 0 (tokens 0..7) and 16 (tokens 8..15)
#pragma unroll
        for (int v = 0; v < 8; ++v) {
            const int m  = v + (half << 3);
            const int bi = best_i[v];
            idx_lds[warp * 16 + m] = bi;
            out_idx[tbase + m]     = (float)bi;
            atomicAdd(&counts[bi], 1.0f);
        }
    }
    __syncthreads();

    // ---- gather quantized output + loss partials (block owns 128 tokens)
    const int blkTokBase = blockIdx.x * 128;
    float acc = 0.0f;
#pragma unroll 4
    for (int e = 0; e < 32; ++e) {
        const int el = e * 256 + tid;       // 0..8191, coalesced
        const int tl = el >> 6;             // local token 0..127
        const int d  = el & 63;
        const int ci = idx_lds[tl];
        const size_t g = (size_t)(blkTokBase + tl) * DIM + d;
        float q  = cb_f32[(size_t)ci * DIM + d];
        float xv = x[g];
        float df = q - xv;
        acc += df * df;
        out_q[g] = q;
    }
#pragma unroll
    for (int off = 16; off > 0; off >>= 1) acc += __shfl_down(acc, off, 32);
    if (lane == 0) red[warp] = acc;
    __syncthreads();
    if (warp == 0) {
        float s = (lane < 8) ? red[lane] : 0.0f;
#pragma unroll
        for (int off = 4; off > 0; off >>= 1) s += __shfl_down(s, off, 32);
        if (lane == 0) atomicAdd(loss, s);
    }
}

// ---------------------------------------------------------------------------
// Finalize: vq_loss = 1.25 * MSE ; perplexity = exp(-sum p*log(p+1e-10))
// ---------------------------------------------------------------------------
__global__ void vq_final(const float* __restrict__ counts,
                         const float* __restrict__ loss,
                         float* __restrict__ out_loss,
                         float* __restrict__ out_perp) {
    __shared__ float red[32];
    const int tid = threadIdx.x, lane = tid & 31, warp = tid >> 5;
    float p = counts[tid] * (1.0f / (float)N_TOK);
    float t = p * logf(p + 1e-10f);
#pragma unroll
    for (int off = 16; off > 0; off >>= 1) t += __shfl_down(t, off, 32);
    if (lane == 0) red[warp] = t;
    __syncthreads();
    if (warp == 0) {
        float s = red[lane];
#pragma unroll
        for (int off = 16; off > 0; off >>= 1) s += __shfl_down(s, off, 32);
        if (lane == 0) {
            *out_perp = expf(-s);
            *out_loss = (*loss) * 1.25f / (float)QUANT_N;
        }
    }
}

// ---------------------------------------------------------------------------
extern "C" void kernel_launch(void* const* d_in, const int* in_sizes, int n_in,
                              void* d_out, int out_size, void* d_ws, size_t ws_size,
                              hipStream_t stream) {
    const float* x  = (const float*)d_in[0];   // [65536, 64]
    const float* cb = (const float*)d_in[1];   // [1024, 64]
    float* out = (float*)d_out;
    char*  ws  = (char*)d_ws;

    __bf16* cb_bf  = (__bf16*)(ws + WS_CBBF);
    float*  esq    = (float*)(ws + WS_ESQ);
    float*  counts = (float*)(ws + WS_CNT);
    float*  loss   = (float*)(ws + WS_LOSS);

    vq_prep<<<4, 256, 0, stream>>>(cb, cb_bf, esq, counts, loss);

    const size_t shmem = 131072 + 4096 + 512 + 32;  // 135,712 B < 320 KB
    vq_main<<<512, 256, shmem, stream>>>(x, cb_bf, esq, cb,
                                         out,                       // quantized
                                         out + QUANT_N + 2,         // indices (as f32)
                                         counts, loss);

    vq_final<<<1, 1024, 0, stream>>>(counts, loss,
                                     out + QUANT_N,                 // vq_loss
                                     out + QUANT_N + 1);            // perplexity
}
